// ContinuousEmbedding_89412629168402
// MI455X (gfx1250) — matrix operand order repaired and checked by reading, not compiled, not run on hardware
//
#include <hip/hip_runtime.h>
#include <hip/hip_bf16.h>

// CDNA5 / gfx1250 wave32 WMMA types
typedef __attribute__((ext_vector_type(16))) __bf16          v16bf;
typedef __attribute__((ext_vector_type(8)))  float           v8f;
typedef __attribute__((ext_vector_type(16))) unsigned short  v16u16;

#define NUM_POINTS 64
#define OUT_DIMS   128
// xs = (x - MINVAL) * NUM_POINTS/(MAX-MIN) = (x + 1) * 32
#define XS_SCALE   32.0f
#define XS_BIAS    1.0f
// Padded LDS row stride (shorts) for the transposed bf16 embedding tile:
// byte stride 132 -> consecutive n hit distinct banks; k-pairs stay 4B-aligned.
#define LDSB_STRIDE 66

__device__ __forceinline__ unsigned bfbits(float f) {
  // round-to-nearest-even f32 -> bf16, returned in low 16 bits
  unsigned u = __builtin_bit_cast(unsigned, f);
  u += 0x7FFFu + ((u >> 16) & 1u);
  return u >> 16;
}

__global__ __launch_bounds__(256)
void ContinuousEmbedding_89412629168402_kernel(const float* __restrict__ x,
                                               const float* __restrict__ emb,
                                               float* __restrict__ out,
                                               int nrows) {
  // Transposed bf16 embeddings: ldsB[n*66 + k] = bf16(emb[k*128 + n])
  __shared__ unsigned short ldsB[OUT_DIMS * LDSB_STRIDE];

  const int tid  = threadIdx.x;
  const int lane = tid & 31;          // wave32
  const int wave = tid >> 5;          // 0..7

  // ---- Stage embeddings (64x128 f32) -> LDS bf16, transposed, packed pairs ----
  // 4096 k-pairs; pair (k,k+1) is adjacent in LDS -> one ds_store_b32 each.
  for (int i = tid; i < (NUM_POINTS / 2) * OUT_DIMS; i += 256) {
    int kp = i >> 7;                 // 0..31  -> k = 2*kp
    int n  = i & 127;                // output dim
    int k  = kp << 1;
    unsigned lo = bfbits(emb[k * OUT_DIMS + n]);
    unsigned hi = bfbits(emb[(k + 1) * OUT_DIMS + n]);
    *(unsigned*)(&ldsB[n * LDSB_STRIDE + k]) = lo | (hi << 16);
  }
  __syncthreads();

  // Wave -> (row tile, N half) mapping: 4 row-tiles x 2 N-halves per block
  const int rowTile = wave >> 1;                 // 0..3, 16 rows each
  const int nbase   = (wave & 1) * 64;           // N half: [0,64) or [64,128)
  const int rowBase = blockIdx.x * 64 + rowTile * 16;

  const int halfSel = lane >> 4;                 // 0: lanes 0-15, 1: lanes 16-31
  const int laneN   = lane & 15;

  // ---- Build B fragments from LDS (lane col = laneN, K = 8*halfSel + (e<8 ? e : e+8)) ----
  v16bf bfrag[4][2];
#pragma unroll
  for (int t = 0; t < 4; ++t) {
#pragma unroll
    for (int c = 0; c < 2; ++c) {
      const int n = nbase + t * 16 + laneN;
      v16u16 bu;
#pragma unroll
      for (int j = 0; j < 8; ++j) {
        const int k0 = c * 32 + halfSel * 8 + 2 * j + ((j < 4) ? 0 : 8);       // even -> 4B aligned
        unsigned p = *(const unsigned*)(&ldsB[n * LDSB_STRIDE + k0]);
        bu[2 * j]     = (unsigned short)p;
        bu[2 * j + 1] = (unsigned short)(p >> 16);
      }
      bfrag[t][c] = __builtin_bit_cast(v16bf, bu);
    }
  }

  // ---- Per-row window weights, exploiting cos periodicity mod 16 ----
  // hann: w(d) = cos^2(pi*d/16)*rect = 0.5 + 0.5*cos(2*pi*(d/16)) for |d|<4.
  // cos(2*pi*(xs-k)/16) depends only on k mod 16; each lane's 32 A-fragment
  // K values cover exactly the 8 residues m = 8*halfSel + 0..7.
  int row = rowBase + laneN;                      // lanes l and l+16 share a row
  int rowClamped = row < nrows ? row : (nrows - 1);
  const float xs = (x[rowClamped] + XS_BIAS) * XS_SCALE;    // grid coords [0,64]

  const int mbase = halfSel * 8;
  float craw[8];                                  // 0.5 + 0.5*cos per residue
#pragma unroll
  for (int i = 0; i < 8; ++i) {
    float m = (float)(mbase + i);
    craw[i] = 0.5f + 0.5f * __builtin_amdgcn_cosf((xs - m) * 0.0625f);
  }

  // Partial row-sum over this lane's 8 residues: the window holds at most one
  // point per residue (width 8 < period 16) -> nearest congruent point test.
  float psum = 0.0f;
#pragma unroll
  for (int i = 0; i < 8; ++i) {
    float m = (float)(mbase + i);
    float q = __builtin_rintf((xs - m) * 0.0625f);          // v_rndne_f32
    float k = __builtin_fmaf(16.0f, q, m);                  // nearest k = m+16q
    float d = xs - k;
    bool in = (__builtin_fabsf(d) < 4.0f) & (k >= 0.0f) & (k <= 63.0f);
    psum += in ? craw[i] : 0.0f;
  }
  // Lanes l and l+16 hold complementary residue sets of the same row: SWAPX16.
  int other = __builtin_amdgcn_ds_swizzle(__builtin_bit_cast(int, psum), 0x401F);
  float s = psum + __builtin_bit_cast(float, other);
  const float inv = (s == 0.0f) ? 0.0f : __builtin_amdgcn_rcpf(s);

  // Pre-pack the 8 normalized residue weights as bf16.
  unsigned short cwbf[8];
#pragma unroll
  for (int i = 0; i < 8; ++i) cwbf[i] = (unsigned short)bfbits(craw[i] * inv);

  // A 16x32 bf16 layout: element e -> K = c*32 + 8*halfSel + (e<8 ? e : e+8),
  // whose residue mod 16 is mbase + (e&7)  -> select cwbf[e&7] under rect.
  v16bf afrag[2];
#pragma unroll
  for (int c = 0; c < 2; ++c) {
    v16u16 au;
#pragma unroll
    for (int e = 0; e < 16; ++e) {
      const int k = c * 32 + mbase + ((e < 8) ? e : e + 8);
      const float d = xs - (float)k;
      au[e] = (__builtin_fabsf(d) < 4.0f) ? cwbf[e & 7] : (unsigned short)0;
    }
    afrag[c] = __builtin_bit_cast(v16bf, au);
  }

  // ---- WMMA: K-chunk outer so same-A ops are adjacent; acc RAW chain is 4
  // instructions apart. EXEC all-1s as required. ----
  v8f acc[4] = {v8f{}, v8f{}, v8f{}, v8f{}};
#pragma unroll
  for (int c = 0; c < 2; ++c) {
#pragma unroll
    for (int t = 0; t < 4; ++t) {
      acc[t] = __builtin_amdgcn_wmma_f32_16x16x32_bf16(
          false, afrag[c], false, bfrag[t][c], (short)0, acc[t], false, false);
    }
  }

  // ---- Store C/D: VGPR r -> row (r + 8*halfSel), col laneN within tile ----
  const int rOut = rowBase + halfSel * 8;
#pragma unroll
  for (int t = 0; t < 4; ++t) {
    const int nOut = nbase + t * 16 + laneN;
#pragma unroll
    for (int r = 0; r < 8; ++r) {
      const int rr = rOut + r;
      if (rr < nrows) out[(size_t)rr * OUT_DIMS + nOut] = acc[t][r];
    }
  }
}

extern "C" void kernel_launch(void* const* d_in, const int* in_sizes, int n_in,
                              void* d_out, int out_size, void* d_ws, size_t ws_size,
                              hipStream_t stream) {
  const float* x   = (const float*)d_in[0];   // (64, 8192) f32, flat 524288
  const float* emb = (const float*)d_in[1];   // (64, 128)  f32
  float* out = (float*)d_out;                 // (64, 8192, 128) f32
  const int nrows = in_sizes[0];              // 524288 rows of the implicit GEMM
  const int blocks = (nrows + 63) / 64;       // 64 rows x 128 cols per block
  ContinuousEmbedding_89412629168402_kernel<<<blocks, 256, 0, stream>>>(x, emb, out, nrows);
}
